// MultiHeadAttention_39444979646622
// MI455X (gfx1250) — compile-verified, matrix operand order checked by reading, not run on hardware
//
#include <hip/hip_runtime.h>
#include <hip/hip_bf16.h>

#define DEV __device__ __forceinline__

typedef __attribute__((ext_vector_type(16))) __bf16 v16bf;
typedef __attribute__((ext_vector_type(8)))  float  v8f;

constexpr int B_  = 2;
constexpr int S_  = 2048;
constexpr int D_  = 1024;
constexpr int H_  = 16;
constexpr int PD_ = 64;
constexpr int ROWS_ = B_ * S_;   // 4096

union FragU { v16bf v; unsigned u[8]; };

// Starting K index of the bf16 pair held in 32-bit reg r, for half hf (lane>>4).
// Matches CDNA5 ISA 16-bit A/B fragment layout: regs 0-3 -> K base 0, regs 4-7 -> K base 16;
// lanes 0-15 hold K-halves {0..7,16..23}, lanes 16-31 hold {8..15,24..31}.
DEV int kpair(int r, int hf) { return ((r & 4) << 2) + hf * 8 + (r & 3) * 2; }

DEV v8f zero8() {
  v8f z;
#pragma unroll
  for (int i = 0; i < 8; ++i) z[i] = 0.0f;
  return z;
}

// Load a 16x32 bf16 fragment (A or B layout) from an LDS tile stored
// row-major [row][K] with even stride; each reg is one aligned 4-byte read
// (compiler merges into ds_load_b128 where the stride allows).
DEV v16bf load_frag(const __bf16* base, int row0, int k0, int stride) {
  const int lane = threadIdx.x & 31;
  const int hf = lane >> 4;
  const __bf16* p = base + (row0 + (lane & 15)) * stride + k0;
  FragU f;
#pragma unroll
  for (int r = 0; r < 8; ++r)
    f.u[r] = *reinterpret_cast<const unsigned*>(p + kpair(r, hf));
  return f.v;
}

DEV v8f wmma_bf16(v16bf a, v16bf b, v8f c) {
  return __builtin_amdgcn_wmma_f32_16x16x32_bf16(false, a, false, b, (short)0, c,
                                                 false, false);
}

// ---------------------------------------------------------------------------
// GEMM: Y = X(MxK) * W(KxN) + bias, M=4096, K=N=1024, via bf16 WMMA, f32 acc.
// HEADSPLIT: write bf16 into [B,H,S,64]; else write f32 into flat [M,N].
// ---------------------------------------------------------------------------
template <typename InT, bool HEADSPLIT>
__global__ __launch_bounds__(256)
void gemm_bias_wmma(const InT* __restrict__ X, const float* __restrict__ W,
                    const float* __restrict__ bias, void* __restrict__ Yout) {
  constexpr int BM = 128, BN = 128, BK = 32, LDA = BK + 8;
  __shared__ __bf16 Xs[BM * LDA];
  __shared__ __bf16 Wt[BN * LDA];   // W tile transposed: [n][k]
  const int tid = threadIdx.x;
  const int lane = tid & 31, wave = tid >> 5, hf = lane >> 4;
  const int m0 = blockIdx.y * BM, n0 = blockIdx.x * BN;
  const int wm = (wave >> 2) * 64, wn = (wave & 3) * 32;

  v8f acc[4][2];
#pragma unroll
  for (int mi = 0; mi < 4; ++mi) { acc[mi][0] = zero8(); acc[mi][1] = zero8(); }

  for (int k0 = 0; k0 < D_; k0 += BK) {
    // Prefetch next tiles into L2 while this tile computes.
    if (k0 + BK < D_) {
      __builtin_prefetch(X + (size_t)(m0 + (tid >> 1)) * D_ + (k0 + BK) + (tid & 1) * 16, 0, 0);
      __builtin_prefetch(W + (size_t)(k0 + BK + (tid >> 3)) * D_ + n0 + (tid & 7) * 16, 0, 0);
    }
    __syncthreads();
    // Stage X tile 128x32 -> bf16 LDS (row-major)
#pragma unroll
    for (int i = 0; i < 4; ++i) {
      int vid = tid + i * 256;
      int r = vid >> 3, c = (vid & 7) << 2;
      __bf16 h[4];
      if constexpr (sizeof(InT) == 4) {
        float4 f = *reinterpret_cast<const float4*>(X + (size_t)(m0 + r) * D_ + k0 + c);
        h[0] = (__bf16)f.x; h[1] = (__bf16)f.y; h[2] = (__bf16)f.z; h[3] = (__bf16)f.w;
      } else {
        const __bf16* s = (const __bf16*)(X + (size_t)(m0 + r) * D_ + k0 + c);
        h[0] = s[0]; h[1] = s[1]; h[2] = s[2]; h[3] = s[3];
      }
      __bf16* dst = &Xs[r * LDA + c];
      dst[0] = h[0]; dst[1] = h[1]; dst[2] = h[2]; dst[3] = h[3];
    }
    // Stage W tile 32x128, transposed into Wt[n][k]
#pragma unroll
    for (int i = 0; i < 4; ++i) {
      int vid = tid + i * 256;
      int kr = vid >> 5, c = (vid & 31) << 2;
      float4 f = *reinterpret_cast<const float4*>(W + (size_t)(k0 + kr) * D_ + n0 + c);
      Wt[(c + 0) * LDA + kr] = (__bf16)f.x;
      Wt[(c + 1) * LDA + kr] = (__bf16)f.y;
      Wt[(c + 2) * LDA + kr] = (__bf16)f.z;
      Wt[(c + 3) * LDA + kr] = (__bf16)f.w;
    }
    __syncthreads();

    v16bf bf[2];
    bf[0] = load_frag(Wt, wn + 0, 0, LDA);
    bf[1] = load_frag(Wt, wn + 16, 0, LDA);
#pragma unroll
    for (int mi = 0; mi < 4; ++mi) {
      v16bf af = load_frag(Xs, wm + mi * 16, 0, LDA);
      acc[mi][0] = wmma_bf16(af, bf[0], acc[mi][0]);
      acc[mi][1] = wmma_bf16(af, bf[1], acc[mi][1]);
    }
  }

#pragma unroll
  for (int mi = 0; mi < 4; ++mi)
#pragma unroll
    for (int ni = 0; ni < 2; ++ni) {
      int gn = n0 + wn + ni * 16 + (lane & 15);
      float bv = bias[gn];
#pragma unroll
      for (int r = 0; r < 8; ++r) {
        int gm = m0 + wm + mi * 16 + r + 8 * hf;
        float val = acc[mi][ni][r] + bv;
        if constexpr (HEADSPLIT) {
          int b = gm >> 11, s = gm & (S_ - 1);
          int h = gn >> 6, pd = gn & (PD_ - 1);
          ((__bf16*)Yout)[(((size_t)b * H_ + h) * S_ + s) * PD_ + pd] = (__bf16)val;
        } else {
          ((float*)Yout)[(size_t)gm * D_ + gn] = val;
        }
      }
    }
}

// ---------------------------------------------------------------------------
// Flash attention: per block = 128 queries of one (batch, head).
// Streams 128-key tiles through LDS; online softmax; WMMA for QK^T and PV.
// K tile is a pure bf16 copy -> use CDNA5 async global->LDS DMA (ASYNCcnt).
// ---------------------------------------------------------------------------
__global__ __launch_bounds__(256)
void attn_wmma(const __bf16* __restrict__ Qh, const __bf16* __restrict__ Kh,
               const __bf16* __restrict__ Vh, __bf16* __restrict__ Ctx,
               const int* __restrict__ use_mask_p) {
  constexpr int QB = 128, KBK = 128;
  constexpr int LDK = PD_ + 8;    // 72 elems = 144 B row stride (16B-aligned)
  constexpr int LDV = KBK + 8;    // 136
  constexpr int LDP = 32 + 8;     // 40
  __shared__ __bf16 Ks[KBK * LDK];   // [key][pd]
  __shared__ __bf16 Vt[PD_ * LDV];   // [pd][key]
  __shared__ __bf16 Ps[8 * 16 * LDP];

  const int tid = threadIdx.x, lane = tid & 31, wave = tid >> 5, hf = lane >> 4;
  const int bh = blockIdx.y;
  const int b = bh >> 4, h = bh & 15;
  const int q0 = blockIdx.x * QB;
  const bool mask = (*use_mask_p) != 0;
  const float MADD_IN  = mask ? -20.0301187f : 0.0f;  // log(2e-9): tril==1
  const float MADD_OUT = mask ? -20.7232658f : 0.0f;  // log(1e-9): tril==0
  const float scale = 0.125f;                          // 1/sqrt(64)

  const __bf16* qbase = Qh + (size_t)bh * S_ * PD_;
  const __bf16* kbase = Kh + (size_t)bh * S_ * PD_;
  const __bf16* vbase = Vh + (size_t)bh * S_ * PD_;

  // Q A-fragments for this wave's 16 rows (held for the whole kernel).
  const int mrow = q0 + wave * 16 + (lane & 15);
  v16bf aq[2];
  {
    FragU f0, f1;
#pragma unroll
    for (int r = 0; r < 8; ++r) {
      f0.u[r] = *reinterpret_cast<const unsigned*>(qbase + (size_t)mrow * PD_ + kpair(r, hf));
      f1.u[r] = *reinterpret_cast<const unsigned*>(qbase + (size_t)mrow * PD_ + 32 + kpair(r, hf));
    }
    aq[0] = f0.v; aq[1] = f1.v;
  }

  v8f o[4];
#pragma unroll
  for (int t = 0; t < 4; ++t) o[t] = zero8();
  float mrun[8], lrun[8];
#pragma unroll
  for (int r = 0; r < 8; ++r) { mrun[r] = -1e30f; lrun[r] = 0.0f; }

  __bf16* Pw = &Ps[wave * 16 * LDP];

  for (int kb = 0; kb < S_; kb += KBK) {
    __syncthreads();
    // K tile: async DMA copy straight into LDS, 16B per lane per issue.
    // LDS byte addr (r*144 + c*2) and global byte addr (r*128 + c*2) are
    // both 16B-aligned for c multiples of 8.
#pragma unroll
    for (int i = 0; i < 4; ++i) {
      int vid = tid + i * 256;
      int r = vid >> 3, c = (vid & 7) << 3;
      unsigned lds_off = (unsigned)(uintptr_t)(&Ks[r * LDK + c]);
      const __bf16* gsrc = kbase + (size_t)(kb + r) * PD_ + c;
      asm volatile("global_load_async_to_lds_b128 %0, %1, off"
                   :: "v"(lds_off), "v"(gsrc)
                   : "memory");
    }
    // V tile: manual transposed staging (layout change; overlaps the DMA).
#pragma unroll
    for (int i = 0; i < 8; ++i) {
      int vid = tid + i * 256;
      int r = vid >> 4, c = (vid & 15) << 2;
      union { unsigned long long u; __bf16 hx[4]; } vv;
      vv.u = *reinterpret_cast<const unsigned long long*>(vbase + (size_t)(kb + r) * PD_ + c);
      Vt[(c + 0) * LDV + r] = vv.hx[0];
      Vt[(c + 1) * LDV + r] = vv.hx[1];
      Vt[(c + 2) * LDV + r] = vv.hx[2];
      Vt[(c + 3) * LDV + r] = vv.hx[3];
    }
    asm volatile("s_wait_asynccnt 0" ::: "memory");
    __syncthreads();

#pragma unroll 1
    for (int ks = 0; ks < KBK; ks += 32) {
      // Logits tile S = Q K^T : 16 x 32 (two 16x16 accumulators).
      v8f sA0 = zero8(), sA1 = zero8();
      {
        v16bf b0 = load_frag(Ks, ks + 0, 0, LDK);
        v16bf b1 = load_frag(Ks, ks + 0, 32, LDK);
        sA0 = wmma_bf16(aq[0], b0, sA0);
        sA0 = wmma_bf16(aq[1], b1, sA0);
        v16bf b2 = load_frag(Ks, ks + 16, 0, LDK);
        v16bf b3 = load_frag(Ks, ks + 16, 32, LDK);
        sA1 = wmma_bf16(aq[0], b2, sA1);
        sA1 = wmma_bf16(aq[1], b3, sA1);
      }
      // Scale + (faithful) mask + online softmax.
#pragma unroll
      for (int r = 0; r < 8; ++r) {
        const int gq  = q0 + wave * 16 + r + 8 * hf;
        const int gk0 = kb + ks + (lane & 15);
        const int gk1 = gk0 + 16;
        float v0 = sA0[r] * scale + ((gk0 <= gq) ? MADD_IN : MADD_OUT);
        float v1 = sA1[r] * scale + ((gk1 <= gq) ? MADD_IN : MADD_OUT);
        float mv = fmaxf(v0, v1);
        mv = fmaxf(mv, __shfl_xor(mv, 1, 32));
        mv = fmaxf(mv, __shfl_xor(mv, 2, 32));
        mv = fmaxf(mv, __shfl_xor(mv, 4, 32));
        mv = fmaxf(mv, __shfl_xor(mv, 8, 32));
        float mnew = fmaxf(mrun[r], mv);
        float alpha = __expf(mrun[r] - mnew);
        mrun[r] = mnew;
        lrun[r] *= alpha;
        o[0][r] *= alpha; o[1][r] *= alpha; o[2][r] *= alpha; o[3][r] *= alpha;
        float e0 = __expf(v0 - mnew);
        float e1 = __expf(v1 - mnew);
        float sum = e0 + e1;
        sum += __shfl_xor(sum, 1, 32);
        sum += __shfl_xor(sum, 2, 32);
        sum += __shfl_xor(sum, 4, 32);
        sum += __shfl_xor(sum, 8, 32);
        lrun[r] += sum;
        // C-layout -> LDS [m][k] (re-layout bounce for the A-fragment of P).
        Pw[(r + 8 * hf) * LDP + (lane & 15)]      = (__bf16)e0;
        Pw[(r + 8 * hf) * LDP + 16 + (lane & 15)] = (__bf16)e1;
      }
      __syncthreads();  // cross-lane LDS ordering (uniform control flow)
      v16bf ap = load_frag(Pw, 0, 0, LDP);
#pragma unroll
      for (int t = 0; t < 4; ++t) {
        v16bf bv = load_frag(Vt, t * 16, ks, LDV);
        o[t] = wmma_bf16(ap, bv, o[t]);
      }
    }
  }

  // Finalize: divide by row sums, store context in flat [B,S,D] (bf16).
  float inv[8];
#pragma unroll
  for (int r = 0; r < 8; ++r) inv[r] = 1.0f / lrun[r];
#pragma unroll
  for (int t = 0; t < 4; ++t)
#pragma unroll
    for (int r = 0; r < 8; ++r) {
      int gq = q0 + wave * 16 + r + 8 * hf;
      int pd = t * 16 + (lane & 15);
      Ctx[((size_t)(b * S_ + gq)) * D_ + h * PD_ + pd] = (__bf16)(o[t][r] * inv[r]);
    }
}

// ---------------------------------------------------------------------------
extern "C" void kernel_launch(void* const* d_in, const int* in_sizes, int n_in,
                              void* d_out, int out_size, void* d_ws, size_t ws_size,
                              hipStream_t stream) {
  (void)in_sizes; (void)n_in; (void)out_size; (void)ws_size;
  const float* q  = (const float*)d_in[0];
  const float* k  = (const float*)d_in[1];
  const float* v  = (const float*)d_in[2];
  const float* Wq = (const float*)d_in[3];
  const float* bq = (const float*)d_in[4];
  const float* Wk = (const float*)d_in[5];
  const float* bk = (const float*)d_in[6];
  const float* Wv = (const float*)d_in[7];
  const float* bv = (const float*)d_in[8];
  const float* Wo = (const float*)d_in[9];
  const float* bo = (const float*)d_in[10];
  const int* use_mask = (const int*)d_in[11];
  float* out = (float*)d_out;

  const size_t n = (size_t)ROWS_ * D_;      // 4M bf16 elems per buffer
  __bf16* Qh  = (__bf16*)d_ws;
  __bf16* Kh  = Qh + n;
  __bf16* Vh  = Kh + n;
  __bf16* Ctx = Vh + n;                      // 32 MB total workspace

  dim3 gGemm(D_ / 128, ROWS_ / 128);         // (8, 32)
  dim3 blk(256);
  gemm_bias_wmma<float, true><<<gGemm, blk, 0, stream>>>(q, Wq, bq, Qh);
  gemm_bias_wmma<float, true><<<gGemm, blk, 0, stream>>>(k, Wk, bk, Kh);
  gemm_bias_wmma<float, true><<<gGemm, blk, 0, stream>>>(v, Wv, bv, Vh);
  dim3 gAttn(S_ / 128, B_ * H_);             // (16, 32)
  attn_wmma<<<gAttn, blk, 0, stream>>>(Qh, Kh, Vh, Ctx, use_mask);
  gemm_bias_wmma<__bf16, false><<<gGemm, blk, 0, stream>>>(Ctx, Wo, bo, out);
}
